// EmpiricalDistribution_56556129354502
// MI455X (gfx1250) — matrix operand order repaired and checked by reading, not compiled, not run on hardware
//
#include <hip/hip_runtime.h>

typedef unsigned int u32;
typedef unsigned short u16;
typedef unsigned long long u64;

#define M_ROWS  4096
#define N_DATA  65536
#define DIM     512
#define NSPLIT  16
#define ROWBLK  64          // x-rows per block = 4 WMMA M-tiles
#define MTILES  4
#define WAVES   16          // 512 threads/block
#define THREADS (WAVES * 32)
#define KCHUNK  64          // K staging chunk for B (double-buffered)
#define BTILE   (16 * KCHUNK)   // 1024 u16 per B buffer
// -( 0.5*512*ln(2*pi) + ln(65536) )
#define KDE_CONST (-481.5868838897515f)

typedef __attribute__((ext_vector_type(16))) __bf16 v16bf;
typedef __attribute__((ext_vector_type(8)))  float  v8f;
typedef __attribute__((ext_vector_type(4)))  u32    u32x4;
typedef __attribute__((ext_vector_type(4)))  int    i32x4;
typedef __attribute__((ext_vector_type(8)))  int    i32x8;

union FragBF { uint4 q[2]; v16bf v; };

__device__ __forceinline__ u32 pk_hi(float a, float b) {
  return (__float_as_uint(a) >> 16) | (__float_as_uint(b) & 0xFFFF0000u);
}
__device__ __forceinline__ float hi_part(float a) {
  return __uint_as_float(__float_as_uint(a) & 0xFFFF0000u);
}
__device__ __forceinline__ void lse_merge(float& mx, float& sm, float omx, float osm) {
  float nm = fmaxf(mx, omx);
  sm = sm * __expf(mx - nm) + osm * __expf(omx - nm);
  mx = nm;
}

#if defined(__has_builtin)
#if __has_builtin(__builtin_amdgcn_tensor_load_to_lds)
#define USE_TDM 1
#endif
#endif

#ifdef USE_TDM
// 2-D tile DMA: global (row-major, stride0 elements of 2B) -> LDS (packed tile rows)
__device__ __forceinline__ void tdm_load_2d(void* ldsp, const void* gp,
                                            u32 tile_d0, u32 tile_d1,
                                            u32 tensor_d0, u32 tensor_d1, u32 stride0) {
  u64 ga = (u64)(uintptr_t)gp;
  u32 la = (u32)(uintptr_t)ldsp;        // low 32 bits of flat LDS addr = LDS byte offset
  u32x4 g0;
  g0[0] = 1u;                            // count=1, user descriptor
  g0[1] = la;                            // lds_addr
  g0[2] = (u32)ga;                       // global_addr[31:0]
  g0[3] = (u32)(ga >> 32) | (2u << 30);  // global_addr[56:32], type=2 (image)
  i32x8 g1;
  g1[0] = 0x00010000;                    // workgroup_mask=0, data_size=1 (2 bytes)
  g1[1] = (int)((tensor_d0 & 0xFFFFu) << 16);                    // tensor_dim0[15:0]
  g1[2] = (int)((tensor_d0 >> 16) | ((tensor_d1 & 0xFFFFu) << 16));
  g1[3] = (int)((tensor_d1 >> 16) | (tile_d0 << 16));            // tile_dim0
  g1[4] = (int)tile_d1;                  // tile_dim1 (tile_dim2 = 0)
  g1[5] = (int)stride0;                  // tensor_dim0_stride[31:0]
  g1[6] = 0;
  g1[7] = 0;
  i32x4 z4 = {0, 0, 0, 0};
#if __clang_major__ >= 23
  i32x8 z8 = {0, 0, 0, 0, 0, 0, 0, 0};
  __builtin_amdgcn_tensor_load_to_lds(g0, g1, z4, z4, z8, 0);
#else
  __builtin_amdgcn_tensor_load_to_lds(g0, g1, z4, z4, 0);
#endif
}
#define WAIT_TENSOR(n) asm volatile("s_wait_tensorcnt " #n ::: "memory")
#else
// Fallback: wave-cooperative copy (DS ops stay in order within the wave)
__device__ __forceinline__ void tdm_load_2d(void* ldsp, const void* gp,
                                            u32 tile_d0, u32 tile_d1,
                                            u32 tensor_d0, u32 tensor_d1, u32 stride0) {
  (void)tensor_d0; (void)tensor_d1;
  const int L = threadIdx.x & 31;
  u16* d = (u16*)ldsp;
  const u16* s = (const u16*)gp;
  const int total8 = (int)(tile_d0 * tile_d1) / 8;
  for (int i = L; i < total8; i += 32) {
    int e = i * 8;
    int r = e / (int)tile_d0, cc = e % (int)tile_d0;
    *(uint4*)(d + e) = *(const uint4*)(s + (size_t)r * stride0 + cc);
  }
}
#define WAIT_TENSOR(n) asm volatile("s_wait_dscnt 0x0" ::: "memory")
#endif

// ---------- Kernel A: split f32 -> bf16 hi/lo (one-time pre-conversion) ------
__global__ void conv_split(const float4* __restrict__ src, uint2* __restrict__ hi,
                           uint2* __restrict__ lo, int n4) {
  int q = blockIdx.x * blockDim.x + threadIdx.x;
  if (q >= n4) return;
  float4 f = src[q];
  uint2 h, l;
  h.x = pk_hi(f.x, f.y); h.y = pk_hi(f.z, f.w);
  float l0 = f.x - hi_part(f.x), l1 = f.y - hi_part(f.y);
  float l2 = f.z - hi_part(f.z), l3 = f.w - hi_part(f.w);
  l.x = pk_hi(l0, l1); l.y = pk_hi(l2, l3);
  hi[q] = h; lo[q] = l;
}

// ---------- Kernel B: dn[j] = 0.5*||data_j||^2 (wave per row) ----------------
__global__ void kde_dnorm(const float* __restrict__ data, float* __restrict__ dn) {
  const int w = threadIdx.x >> 5, L = threadIdx.x & 31;
  const int row = blockIdx.x * 8 + w;
  const float* p = data + (size_t)row * DIM;
  float s = 0.f;
#pragma unroll
  for (int i = 0; i < DIM / 32; ++i) { float v = p[L + 32 * i]; s = fmaf(v, v, s); }
#pragma unroll
  for (int m = 16; m >= 1; m >>= 1) s += __shfl_xor(s, m, 32);
  if (L == 0) dn[row] = 0.5f * s;
}

// ---------- Kernel C: TDM-fed split-bf16 WMMA GEMM + online logsumexp --------
__global__ __launch_bounds__(THREADS, 1)
void kde_main(const u16* __restrict__ xh, const u16* __restrict__ xl,
              const u16* __restrict__ dh, const u16* __restrict__ dl,
              const float* __restrict__ dn, float2* __restrict__ partial) {
  extern __shared__ char smem[];
  u16* Ah = (u16*)smem;                               // [ROWBLK][DIM]
  u16* Al = Ah + ROWBLK * DIM;
  u16* Bh = Al + ROWBLK * DIM;                        // [WAVES][2][BTILE]
  u16* Bl = Bh + WAVES * 2 * BTILE;
  float2* red = (float2*)(Bl + WAVES * 2 * BTILE);    // [ROWBLK][WAVES]

  const int tid = threadIdx.x;
  const int w = tid >> 5, L = tid & 31;
  const int g = L >> 4, n = L & 15;
  const int rb = blockIdx.x / NSPLIT;
  const int ns = blockIdx.x % NSPLIT;

  // ---- stage A tile via TDM: each wave DMAs its 4-row hi/lo slice ----
  tdm_load_2d(Ah + w * 4 * DIM, xh + ((size_t)rb * ROWBLK + w * 4) * DIM,
              DIM, 4, DIM, M_ROWS, DIM);
  tdm_load_2d(Al + w * 4 * DIM, xl + ((size_t)rb * ROWBLK + w * 4) * DIM,
              DIM, 4, DIM, M_ROWS, DIM);
  WAIT_TENSOR(0x0);
  __syncthreads();

  float mx[MTILES][8], sm[MTILES][8];
#pragma unroll
  for (int t = 0; t < MTILES; ++t)
#pragma unroll
    for (int v = 0; v < 8; ++v) { mx[t][v] = -3e38f; sm[t][v] = 0.f; }

  const int TOT = ((N_DATA / NSPLIT) / (WAVES * 16)) * (DIM / KCHUNK);   // 128 chunks
  {   // preload chunk 0 into buffer 0
    const int j0 = ns * (N_DATA / NSPLIT) + w * 16;
    tdm_load_2d(Bh + (w * 2 + 0) * BTILE, dh + (size_t)j0 * DIM, KCHUNK, 16, DIM, N_DATA, DIM);
    tdm_load_2d(Bl + (w * 2 + 0) * BTILE, dl + (size_t)j0 * DIM, KCHUNK, 16, DIM, N_DATA, DIM);
  }

  float dnj = 0.f;
  v8f c[MTILES];
  for (int ci = 0; ci < TOT; ++ci) {
    const int it = ci >> 3, kc = ci & 7;
    const int j0 = ns * (N_DATA / NSPLIT) + it * (WAVES * 16) + w * 16;
    const int buf = ci & 1;

    if (ci + 1 < TOT) {
      const int itn = (ci + 1) >> 3, kcn = (ci + 1) & 7;
      const int jn = ns * (N_DATA / NSPLIT) + itn * (WAVES * 16) + w * 16;
      // buffer buf^1 was last read by ds_loads of iteration ci-1: drain them first
      asm volatile("s_wait_dscnt 0x0" ::: "memory");
      tdm_load_2d(Bh + (w * 2 + (buf ^ 1)) * BTILE, dh + (size_t)jn * DIM + kcn * KCHUNK,
                  KCHUNK, 16, DIM, N_DATA, DIM);
      tdm_load_2d(Bl + (w * 2 + (buf ^ 1)) * BTILE, dl + (size_t)jn * DIM + kcn * KCHUNK,
                  KCHUNK, 16, DIM, N_DATA, DIM);
      WAIT_TENSOR(0x2);        // chunk ci's pair complete; ci+1's pair in flight
    } else {
      WAIT_TENSOR(0x0);
    }

    if (kc == 0) {
      dnj = dn[j0 + n];
#pragma unroll
      for (int t = 0; t < MTILES; ++t)
#pragma unroll
        for (int e = 0; e < 8; ++e) c[t][e] = 0.f;
    }

    // NOTE: keep these as single expressions rooted at smem so addrspace
    // inference yields ds_load_b128 (an array-of-pointers select lowered to
    // v_cndmask on generic pointers and produced flat_load_b128 last round).
    const int bbase = (w * 2 + buf) * BTILE;
#pragma unroll
    for (int ks = 0; ks < KCHUNK / 32; ++ks) {
      const int kkB = ks * 32 + g * 16;
      FragBF bh, bl;
      bh.q[0] = *(const uint4*)&Bh[bbase + n * KCHUNK + kkB];
      bh.q[1] = *(const uint4*)&Bh[bbase + n * KCHUNK + kkB + 8];
      bl.q[0] = *(const uint4*)&Bl[bbase + n * KCHUNK + kkB];
      bl.q[1] = *(const uint4*)&Bl[bbase + n * KCHUNK + kkB + 8];
      const int kgA = kc * KCHUNK + ks * 32 + g * 16;
#pragma unroll
      for (int t = 0; t < MTILES; ++t) {
        const int m = t * 16 + n;
        FragBF ah, al;
        ah.q[0] = *(const uint4*)&Ah[m * DIM + kgA];
        ah.q[1] = *(const uint4*)&Ah[m * DIM + kgA + 8];
        al.q[0] = *(const uint4*)&Al[m * DIM + kgA];
        al.q[1] = *(const uint4*)&Al[m * DIM + kgA + 8];
        c[t] = __builtin_amdgcn_wmma_f32_16x16x32_bf16(false, ah.v, false, bh.v,
                                                       (short)0, c[t], false, false);
        c[t] = __builtin_amdgcn_wmma_f32_16x16x32_bf16(false, ah.v, false, bl.v,
                                                       (short)0, c[t], false, false);
        c[t] = __builtin_amdgcn_wmma_f32_16x16x32_bf16(false, al.v, false, bh.v,
                                                       (short)0, c[t], false, false);
      }
    }

    if (kc == 7) {   // full K accumulated for this 16-col tile: online LSE
#pragma unroll
      for (int t = 0; t < MTILES; ++t)
#pragma unroll
        for (int v = 0; v < 8; ++v) {
          float e = c[t][v] - dnj;
          lse_merge(mx[t][v], sm[t][v], e, 1.0f);
        }
    }
  }

  // ---- combine across the 16 lanes of each column group ----
#pragma unroll
  for (int t = 0; t < MTILES; ++t)
#pragma unroll
    for (int v = 0; v < 8; ++v)
#pragma unroll
      for (int m = 1; m < 16; m <<= 1) {
        float omx = __shfl_xor(mx[t][v], m, 32);
        float osm = __shfl_xor(sm[t][v], m, 32);
        lse_merge(mx[t][v], sm[t][v], omx, osm);
      }
  if (n == 0) {
#pragma unroll
    for (int t = 0; t < MTILES; ++t)
#pragma unroll
      for (int v = 0; v < 8; ++v)
        red[(t * 16 + g * 8 + v) * WAVES + w] = make_float2(mx[t][v], sm[t][v]);
  }
  __syncthreads();

  if (tid < ROWBLK) {
    float M = -3e38f, S = 0.f;
#pragma unroll
    for (int ww = 0; ww < WAVES; ++ww) {
      float2 p = red[tid * WAVES + ww];
      lse_merge(M, S, p.x, p.y);
    }
    partial[(size_t)(rb * ROWBLK + tid) * NSPLIT + ns] = make_float2(M, S);
  }
}

// ---------- Kernel D: final reduce + constants (wave per row) ----------------
__global__ void kde_reduce(const float* __restrict__ x,
                           const float2* __restrict__ partial,
                           float* __restrict__ out) {
  const int w = threadIdx.x >> 5, L = threadIdx.x & 31;
  const int row = blockIdx.x * 8 + w;
  const float* p = x + (size_t)row * DIM;
  float s = 0.f;
#pragma unroll
  for (int i = 0; i < DIM / 32; ++i) { float v = p[L + 32 * i]; s = fmaf(v, v, s); }
#pragma unroll
  for (int m = 16; m >= 1; m >>= 1) s += __shfl_xor(s, m, 32);

  float M = -3e38f, S = 0.f;
  if (L < NSPLIT) { float2 q = partial[(size_t)row * NSPLIT + L]; M = q.x; S = q.y; }
#pragma unroll
  for (int m = 1; m < 16; m <<= 1) {
    float omx = __shfl_xor(M, m, 32);
    float osm = __shfl_xor(S, m, 32);
    lse_merge(M, S, omx, osm);
  }
  if (L == 0) out[row] = KDE_CONST + M + __logf(S) - 0.5f * s;
}

extern "C" void kernel_launch(void* const* d_in, const int* in_sizes, int n_in,
                              void* d_out, int out_size, void* d_ws, size_t ws_size,
                              hipStream_t stream) {
  (void)in_sizes; (void)n_in; (void)out_size; (void)ws_size;
  const float* x    = (const float*)d_in[0];
  const float* data = (const float*)d_in[1];
  float* out = (float*)d_out;

  // workspace layout
  char* ws = (char*)d_ws;
  float*  dn      = (float*)ws;                                   // 256 KB
  float2* partial = (float2*)(ws + (size_t)N_DATA * 4);           // 512 KB
  u16* xh = (u16*)(ws + (size_t)N_DATA * 4 + (size_t)M_ROWS * NSPLIT * 8);
  u16* xl = xh + (size_t)M_ROWS * DIM;                            // 4 MB each
  u16* dh = xl + (size_t)M_ROWS * DIM;                            // 64 MB each
  u16* dl = dh + (size_t)N_DATA * DIM;

  // one-time split conversion f32 -> bf16 hi/lo
  conv_split<<<(M_ROWS * DIM / 4) / 256, 256, 0, stream>>>(
      (const float4*)x, (uint2*)xh, (uint2*)xl, M_ROWS * DIM / 4);
  conv_split<<<(N_DATA * DIM / 4) / 256, 256, 0, stream>>>(
      (const float4*)data, (uint2*)dh, (uint2*)dl, N_DATA * DIM / 4);

  kde_dnorm<<<N_DATA / 8, 256, 0, stream>>>(data, dn);

  const size_t smem = (size_t)2 * ROWBLK * DIM * 2                // A hi/lo
                    + (size_t)2 * WAVES * 2 * BTILE * 2           // B hi/lo double-buffered
                    + (size_t)ROWBLK * WAVES * 8;                 // reduction buffer
  kde_main<<<(M_ROWS / ROWBLK) * NSPLIT, THREADS, smem, stream>>>(xh, xl, dh, dl, dn, partial);

  kde_reduce<<<M_ROWS / 8, 256, 0, stream>>>(x, partial, out);
}
